// SelfAttention_83399674954166
// MI455X (gfx1250) — compile-verified, matrix operand order checked by reading, not run on hardware
//
#include <hip/hip_runtime.h>

// ---------------------------------------------------------------------------
// Self-attention (B=2, S=2048, D=1024, H=16, Dh=64) for MI455X (gfx1250).
// All matmuls via v_wmma_f32_16x16x32_bf16 (wave32), softmax/accum in fp32.
// GEMMs use 64x32 per-wave register blocking (4x2 fragments, 8 WMMA/k-step)
// -- sized to stay inside the 256-VGPR direct window (no scratch spills).
// ---------------------------------------------------------------------------

typedef __bf16  v16bf __attribute__((ext_vector_type(16)));
typedef __bf16  v8bf  __attribute__((ext_vector_type(8)));
typedef float   v8f   __attribute__((ext_vector_type(8)));

#define B_     2
#define S_     2048
#define D_     1024
#define H_     16
#define DH_    64
#define M_TOK  (B_ * S_)        // 4096 token rows
#define N_QKV  (3 * D_)         // 3072

__device__ __forceinline__ v8f wmma_bf16(v16bf a, v16bf b, v8f c) {
    // 8 args: (neg_a, A, neg_b, B, c_mod, C, reuse_a, reuse_b)
    return __builtin_amdgcn_wmma_f32_16x16x32_bf16(
        false, a, false, b, (short)0, c, false, false);
}

// Load one 16x32 bf16 fragment (A row-major, or B with n-major storage).
// base points at element (0,0) of the tile; ld = leading dimension in elems.
// Lane layout (ISA 7.12.2, 16-bit A): r = lane&15 (row M / col N),
// kg = lane>>4; elems 0..7 -> K = kg*8 + 0..7 ; elems 8..15 -> K = 16+kg*8+0..7
__device__ __forceinline__ v16bf load_frag(const __bf16* base, int ld) {
    int lane = threadIdx.x & 31;
    int r    = lane & 15;
    int kg   = lane >> 4;
    const __bf16* p0 = base + (size_t)r * ld + kg * 8;
    v8bf lo = *(const v8bf*)(p0);
    v8bf hi = *(const v8bf*)(p0 + 16);
    return __builtin_shufflevector(lo, hi,
        0, 1, 2, 3, 4, 5, 6, 7, 8, 9, 10, 11, 12, 13, 14, 15);
}

// Reductions within each 16-lane half (masks < 16 never cross halves).
__device__ __forceinline__ float redmax16(float v) {
    v = fmaxf(v, __shfl_xor(v, 1, 32));
    v = fmaxf(v, __shfl_xor(v, 2, 32));
    v = fmaxf(v, __shfl_xor(v, 4, 32));
    v = fmaxf(v, __shfl_xor(v, 8, 32));
    return v;
}
__device__ __forceinline__ float redsum16(float v) {
    v += __shfl_xor(v, 1, 32);
    v += __shfl_xor(v, 2, 32);
    v += __shfl_xor(v, 4, 32);
    v += __shfl_xor(v, 8, 32);
    return v;
}

// ---------------------------------------------------------------------------
// Kernel 0: fp32 -> bf16 convert
// ---------------------------------------------------------------------------
__global__ void cvt_f32_bf16(const float* __restrict__ in,
                             __bf16* __restrict__ out, int n) {
    int i = blockIdx.x * blockDim.x + threadIdx.x;
    if (i < n) out[i] = (__bf16)in[i];
}

// ---------------------------------------------------------------------------
// Kernel 1: QKV projection.  qkv[t,o] = sum_d xb[t,d]*w_in[o,d] + b_in[o]
// One wave per 64x32 output block: 4x2 C fragments, 8 WMMA per k-step of 32.
// Scatter: q (scaled by 1/8) -> [B,H,S,Dh], k -> [B,H,S,Dh], v -> [B,H,Dh,S].
// ---------------------------------------------------------------------------
__global__ void __launch_bounds__(256, 1)
gemm_qkv(const __bf16* __restrict__ xb,
         const __bf16* __restrict__ wb,   // [3D, D] row-major
         const float*  __restrict__ b_in,
         __bf16* __restrict__ qbuf,
         __bf16* __restrict__ kbuf,
         __bf16* __restrict__ vT) {
    const int NT = N_QKV / 32;                      // 96 column blocks
    int gwave = blockIdx.x * (blockDim.x >> 5) + (threadIdx.x >> 5);
    if (gwave >= (M_TOK / 64) * NT) return;         // wave-uniform
    int mt = gwave / NT, nt = gwave % NT;
    int row0 = mt * 64, col0 = nt * 32;

    const __bf16* arow = xb + (size_t)row0 * D_;
    const __bf16* brow = wb + (size_t)col0 * D_;

    v8f acc[4][2] = {};
    for (int k = 0; k < D_; k += 32) {
        __builtin_prefetch(arow + k + 32, 0, 1);    // global_prefetch_b8
        __builtin_prefetch(brow + k + 32, 0, 1);
        v16bf af[4], bfr[2];
        #pragma unroll
        for (int i = 0; i < 4; ++i)
            af[i] = load_frag(arow + (size_t)(i * 16) * D_ + k, D_);
        #pragma unroll
        for (int j = 0; j < 2; ++j)
            bfr[j] = load_frag(brow + (size_t)(j * 16) * D_ + k, D_);
        #pragma unroll
        for (int i = 0; i < 4; ++i)
            #pragma unroll
            for (int j = 0; j < 2; ++j)
                acc[i][j] = wmma_bf16(af[i], bfr[j], acc[i][j]);
    }

    int lane = threadIdx.x & 31;
    int lanelo = lane & 15, half = lane >> 4;
    #pragma unroll
    for (int j = 0; j < 2; ++j) {
        int o = col0 + j * 16 + lanelo;
        float bias = b_in[o];
        int oq = o & (D_ - 1);
        int h = oq >> 6, d = oq & 63;
        #pragma unroll
        for (int i = 0; i < 4; ++i) {
            #pragma unroll
            for (int r = 0; r < 8; ++r) {
                int tok = row0 + i * 16 + r + (half << 3);  // M = r + 8*half
                int b_  = tok >> 11;                        // /2048
                int s_  = tok & (S_ - 1);
                float val = acc[i][j][r] + bias;
                size_t bh = (size_t)(b_ * H_ + h);
                if (o < D_) {                               // Q (pre-scaled)
                    qbuf[(bh * S_ + s_) * DH_ + d] = (__bf16)(val * 0.125f);
                } else if (o < 2 * D_) {                    // K
                    kbuf[(bh * S_ + s_) * DH_ + d] = (__bf16)val;
                } else {                                    // V, transposed
                    vT[(bh * DH_ + d) * S_ + s_] = (__bf16)val;
                }
            }
        }
    }
}

// ---------------------------------------------------------------------------
// Kernel 2: flash attention.  One wave = one (b,h) x 16-query tile.
// Per 32-key chunk: 4 WMMAs (scores), online softmax, P via LDS re-layout,
// 4 WMMAs (P*V).  Output bf16 at [B,H,S,Dh] flat (== reference reshape).
// ---------------------------------------------------------------------------
__global__ void attn_fwd(const __bf16* __restrict__ qbuf,
                         const __bf16* __restrict__ kbuf,
                         const __bf16* __restrict__ vT,
                         __bf16* __restrict__ attn) {
    __shared__ __bf16 pLds[8 * 16 * 32];            // 1KB per wave, 8 waves
    int wave = threadIdx.x >> 5;
    int lane = threadIdx.x & 31;
    int lanelo = lane & 15, half = lane >> 4;
    int gwave = blockIdx.x * (blockDim.x >> 5) + wave;
    if (gwave >= B_ * H_ * (S_ / 16)) return;       // wave-uniform
    int bh = gwave >> 7;                            // / (S/16)
    int q0 = (gwave & 127) * 16;
    __bf16* myP = pLds + wave * 512;

    const __bf16* qbase = qbuf + ((size_t)bh * S_ + q0) * DH_;
    v16bf qa0 = load_frag(qbase, DH_);              // d = 0..31
    v16bf qa1 = load_frag(qbase + 32, DH_);         // d = 32..63

    float mi[8], li[8];
    v8f oacc[4] = {};
    #pragma unroll
    for (int r = 0; r < 8; ++r) { mi[r] = -1e30f; li[r] = 0.0f; }

    const __bf16* kbh = kbuf + (size_t)bh * S_ * DH_;
    const __bf16* vbh = vT + (size_t)bh * DH_ * S_;

    for (int jc = 0; jc < S_; jc += 32) {
        // --- scores: S[16q x 32k] = Q(16x64) * K^T ---
        const __bf16* kb = kbh + (size_t)jc * DH_;
        __builtin_prefetch(kb + 32 * DH_, 0, 1);    // next chunk of K
        v16bf kf00 = load_frag(kb, DH_);
        v16bf kf01 = load_frag(kb + 32, DH_);
        v16bf kf10 = load_frag(kb + 16 * DH_, DH_);
        v16bf kf11 = load_frag(kb + 16 * DH_ + 32, DH_);
        v8f s0 = {}, s1 = {};
        s0 = wmma_bf16(qa0, kf00, s0);
        s0 = wmma_bf16(qa1, kf01, s0);              // keys jc .. jc+15
        s1 = wmma_bf16(qa0, kf10, s1);
        s1 = wmma_bf16(qa1, kf11, s1);              // keys jc+16 .. jc+31

        // --- online softmax (rows live in VGPR r, 16 cols across half) ---
        float alpha[8];
        #pragma unroll
        for (int r = 0; r < 8; ++r) {
            float mx = redmax16(fmaxf(s0[r], s1[r]));
            float mn = fmaxf(mi[r], mx);
            alpha[r] = __expf(mi[r] - mn);
            mi[r] = mn;
            float p0 = __expf(s0[r] - mn);
            float p1 = __expf(s1[r] - mn);
            li[r] = li[r] * alpha[r] + redsum16(p0 + p1);
            int mrow = r + (half << 3);
            myP[mrow * 32 + lanelo]      = (__bf16)p0;
            myP[mrow * 32 + 16 + lanelo] = (__bf16)p1;
        }
        #pragma unroll
        for (int t = 0; t < 4; ++t)
            #pragma unroll
            for (int r = 0; r < 8; ++r) oacc[t][r] *= alpha[r];

        // --- P(16x32, A layout from LDS) * V(32k x 64d) ---
        v16bf pa = load_frag(myP, 32);
        const __bf16* vb = vbh + jc;                // n-major, ld = S
        #pragma unroll
        for (int t = 0; t < 4; ++t) {
            v16bf vf = load_frag(vb + (size_t)(t * 16) * S_, S_);
            oacc[t] = wmma_bf16(pa, vf, oacc[t]);
        }
    }

    // --- normalize + store bf16 ---
    #pragma unroll
    for (int t = 0; t < 4; ++t) {
        #pragma unroll
        for (int r = 0; r < 8; ++r) {
            int mrow = r + (half << 3);
            float val = oacc[t][r] / li[r];
            attn[((size_t)bh * S_ + q0 + mrow) * DH_ + t * 16 + lanelo] =
                (__bf16)val;
        }
    }
}

// ---------------------------------------------------------------------------
// Kernel 3: output projection.  y[t,o] = sum_d attn[t,d]*w_out[o,d] + b_out[o]
// attn flat [B,H,S,Dh] == [4096, 1024] row-major (reference's reshape).
// 64x32 per-wave register blocking like gemm_qkv; fp32 output + bias.
// ---------------------------------------------------------------------------
__global__ void __launch_bounds__(256, 1)
gemm_out(const __bf16* __restrict__ attn,
         const __bf16* __restrict__ wb,   // [D, D] row-major
         const float*  __restrict__ b_out,
         float* __restrict__ out) {
    const int NT = D_ / 32;                         // 32 column blocks
    int gwave = blockIdx.x * (blockDim.x >> 5) + (threadIdx.x >> 5);
    if (gwave >= (M_TOK / 64) * NT) return;
    int mt = gwave / NT, nt = gwave % NT;
    int row0 = mt * 64, col0 = nt * 32;

    const __bf16* arow = attn + (size_t)row0 * D_;
    const __bf16* brow = wb + (size_t)col0 * D_;

    v8f acc[4][2] = {};
    for (int k = 0; k < D_; k += 32) {
        __builtin_prefetch(arow + k + 32, 0, 1);
        __builtin_prefetch(brow + k + 32, 0, 1);
        v16bf af[4], bfr[2];
        #pragma unroll
        for (int i = 0; i < 4; ++i)
            af[i] = load_frag(arow + (size_t)(i * 16) * D_ + k, D_);
        #pragma unroll
        for (int j = 0; j < 2; ++j)
            bfr[j] = load_frag(brow + (size_t)(j * 16) * D_ + k, D_);
        #pragma unroll
        for (int i = 0; i < 4; ++i)
            #pragma unroll
            for (int j = 0; j < 2; ++j)
                acc[i][j] = wmma_bf16(af[i], bfr[j], acc[i][j]);
    }

    int lane = threadIdx.x & 31;
    int lanelo = lane & 15, half = lane >> 4;
    #pragma unroll
    for (int j = 0; j < 2; ++j) {
        int o = col0 + j * 16 + lanelo;
        float bias = b_out[o];
        #pragma unroll
        for (int i = 0; i < 4; ++i) {
            #pragma unroll
            for (int r = 0; r < 8; ++r) {
                int mrow = row0 + i * 16 + r + (half << 3);
                out[(size_t)mrow * D_ + o] = acc[i][j][r] + bias;
            }
        }
    }
}

// ---------------------------------------------------------------------------
// Host launch
// ---------------------------------------------------------------------------
extern "C" void kernel_launch(void* const* d_in, const int* in_sizes, int n_in,
                              void* d_out, int out_size, void* d_ws, size_t ws_size,
                              hipStream_t stream) {
    const float* x     = (const float*)d_in[0];   // [2,2048,1024]
    const float* w_in  = (const float*)d_in[1];   // [3072,1024]
    const float* b_in  = (const float*)d_in[2];   // [3072]
    const float* w_out = (const float*)d_in[3];   // [1024,1024]
    const float* b_out = (const float*)d_in[4];   // [1024]
    float* out = (float*)d_out;

    // Workspace layout (bf16 buffers, 256B-aligned offsets)
    char* ws = (char*)d_ws;
    __bf16* xb      = (__bf16*)(ws);                              // 8 MB
    __bf16* w_in_b  = (__bf16*)(ws + (size_t)8 * 1024 * 1024);    // 6 MB
    __bf16* w_out_b = (__bf16*)(ws + (size_t)14 * 1024 * 1024);   // 2 MB
    __bf16* qbuf    = (__bf16*)(ws + (size_t)16 * 1024 * 1024);   // 8 MB
    __bf16* kbuf    = (__bf16*)(ws + (size_t)24 * 1024 * 1024);   // 8 MB
    __bf16* vT      = (__bf16*)(ws + (size_t)32 * 1024 * 1024);   // 8 MB
    __bf16* attn    = (__bf16*)(ws + (size_t)40 * 1024 * 1024);   // 8 MB

    // 1) fp32 -> bf16 converts
    {
        int n = M_TOK * D_;                         // 4,194,304
        cvt_f32_bf16<<<(n + 255) / 256, 256, 0, stream>>>(x, xb, n);
        n = N_QKV * D_;                             // 3,145,728
        cvt_f32_bf16<<<(n + 255) / 256, 256, 0, stream>>>(w_in, w_in_b, n);
        n = D_ * D_;                                // 1,048,576
        cvt_f32_bf16<<<(n + 255) / 256, 256, 0, stream>>>(w_out, w_out_b, n);
    }

    // 2) QKV projection: (4096/64)*(3072/32) = 6144 wave-blocks, 8 waves/block
    gemm_qkv<<<6144 / 8, 256, 0, stream>>>(xb, w_in_b, b_in, qbuf, kbuf, vT);

    // 3) attention: 2*16*128 = 4096 waves, 8 waves/block
    attn_fwd<<<4096 / 8, 256, 0, stream>>>(qbuf, kbuf, vT, attn);

    // 4) output projection: (4096/64)*(1024/32) = 2048 wave-blocks
    gemm_out<<<2048 / 8, 256, 0, stream>>>(attn, w_out_b, b_out, out);
}
